// GraphConvTest_16037407883636
// MI455X (gfx1250) — compile-verified, matrix operand order checked by reading, not run on hardware
//
#include <hip/hip_runtime.h>
#include <math.h>

// ---------------------------------------------------------------------------
// Shapes
// ---------------------------------------------------------------------------
#define B_ 32
#define N_ 128
#define F_ 128
#define E_ 8
#define U_ 128
#define UE_ 64
#define K6 96   // 64 (relu(P+Q)) + 8 (lcq_adj) padded to 96 = 3 * 32

typedef __bf16 bf16_t;
typedef bf16_t v16bf __attribute__((ext_vector_type(16)));
typedef float  v8f   __attribute__((ext_vector_type(8)));

// ---------------------------------------------------------------------------
// WMMA helpers (v_wmma_f32_16x16x32_bf16, D = A(16x32) * B(32x16) + C)
// ---------------------------------------------------------------------------
__device__ __forceinline__ v8f wmma_bf16(v16bf a, v16bf b, v8f c) {
  // (neg_a, A, neg_b, B, c_mod, C, reuse_a, reuse_b)
  return __builtin_amdgcn_wmma_f32_16x16x32_bf16(false, a, false, b, (short)0, c,
                                                 false, false);
}

// A / B fragment load from a ROW-MAJOR bf16 matrix (B must be stored as B^T).
// ISA layout (16-bit A 16x32): lane l holds row (l&15); K = (l>>4)*8 + t for
// vector elems 0..7 and K = 16 + (l>>4)*8 + t for elems 8..15 -> two
// contiguous 16-byte chunks per lane.
__device__ __forceinline__ v16bf load_frag_rm(const bf16_t* base, int ld, int lane) {
  const int r  = lane & 15;
  const int kk = (lane >> 4) << 3;
  const bf16_t* p = base + r * ld + kk;
  union { uint4 q[2]; v16bf v; } u;
  u.q[0] = *(const uint4*)(p);
  u.q[1] = *(const uint4*)(p + 16);
  return u.v;
}

// Same fragment, sourced from a row-major f32 matrix (convert on the fly).
__device__ __forceinline__ v16bf load_frag_f32rm(const float* base, int ld, int lane) {
  const int r  = lane & 15;
  const int kk = (lane >> 4) << 3;
  const float* p = base + r * ld + kk;
  v16bf a;
#pragma unroll
  for (int t = 0; t < 8; ++t) a[t] = (bf16_t)p[t];
#pragma unroll
  for (int t = 0; t < 8; ++t) a[8 + t] = (bf16_t)p[16 + t];
  return a;
}

// LDS byte offset of a generic pointer known to alias __shared__ memory.
__device__ __forceinline__ unsigned lds_addr_of(const void* p) {
  return (unsigned)(unsigned long long)
      (__attribute__((address_space(3))) const void*)p;
}

// ---------------------------------------------------------------------------
// Workspace layout (bytes)
// ---------------------------------------------------------------------------
#define WS_SUPPORT   0                        // f32  [B][N][F]      2 MiB
#define WS_SUPPORT_T (2097152)                // bf16 [B][F][N]      1 MiB
#define WS_PQ        (WS_SUPPORT_T + 1048576) // f32  [B][N][128]    2 MiB (P|Q)
#define WS_TESTWT    (WS_PQ + 2097152)        // bf16 [128][128]
#define WS_OUTWT     (WS_TESTWT + 32768)      // bf16 [128][1024]
#define WS_WCATT     (WS_OUTWT + 262144)      // bf16 [128][128]
#define WS_W6T       (WS_WCATT + 32768)       // bf16 [64][96]

// ---------------------------------------------------------------------------
// Kernel 0: transpose/convert all weights to bf16 B^T layouts.
// ---------------------------------------------------------------------------
__global__ void prep_weights(const float* __restrict__ testW,
                             const float* __restrict__ outW,
                             const float* __restrict__ adjW,
                             const float* __restrict__ adjOutW,
                             bf16_t* __restrict__ testWt,
                             bf16_t* __restrict__ outWt,
                             bf16_t* __restrict__ WcatT,
                             bf16_t* __restrict__ W6t) {
  int i = blockIdx.x * blockDim.x + threadIdx.x;
  if (i < F_ * F_) {                       // testWt[n][k] = testW[k][n]
    int n = i >> 7, k = i & 127;
    testWt[n * F_ + k] = (bf16_t)testW[k * F_ + n];
    return;
  }
  i -= F_ * F_;
  if (i < U_ * (F_ * E_)) {                // outWt[u][k] = outW[k][u]
    int u = i >> 10, k = i & 1023;
    outWt[u * (F_ * E_) + k] = (bf16_t)outW[k * U_ + u];
    return;
  }
  i -= U_ * (F_ * E_);
  if (i < 128 * 128) {                     // WcatT[n][k]: n<64 -> W_top, else W_bot
    int n = i >> 7, k = i & 127;
    float v = (n < UE_) ? adjW[k * UE_ + n] : adjW[(U_ + k) * UE_ + (n - UE_)];
    WcatT[n * 128 + k] = (bf16_t)v;
    return;
  }
  i -= 128 * 128;
  if (i < UE_ * K6) {                      // W6t[n][k] = adjOutW[k][n], zero-padded K 72->96
    int n = i / K6, k = i % K6;
    W6t[n * K6 + k] = (k < (UE_ + E_)) ? (bf16_t)adjOutW[k * UE_ + n] : (bf16_t)0.0f;
    return;
  }
}

// ---------------------------------------------------------------------------
// Kernel 1: support = node_feature @ test_W   (M=B*N, K=F, N=F)
// 256 blocks, 8 waves; block = (b, 16-row tile), wave = 16-col tile.
// Writes support (f32) and supportT (bf16, [b][f][n]) for later B-frags.
// ---------------------------------------------------------------------------
__global__ __launch_bounds__(256)
void support_kernel(const float* __restrict__ nf,
                    const bf16_t* __restrict__ testWt,
                    float* __restrict__ support,
                    bf16_t* __restrict__ supportT) {
  const int b    = blockIdx.x >> 3;
  const int n0   = (blockIdx.x & 7) << 4;
  const int wave = threadIdx.x >> 5;
  const int lane = threadIdx.x & 31;
  const int f0   = wave << 4;

  v8f acc = {};
  const float* abase = nf + (b * N_ + n0) * F_;
#pragma unroll
  for (int k0 = 0; k0 < F_; k0 += 32) {
    v16bf a  = load_frag_f32rm(abase + k0, F_, lane);
    v16bf bf = load_frag_rm(testWt + f0 * F_ + k0, F_, lane);
    acc = wmma_bf16(a, bf, acc);
  }
  const int col = f0 + (lane & 15);
  const int mhi = (lane >> 4) << 3;
#pragma unroll
  for (int v = 0; v < 8; ++v) {
    const int row = n0 + v + mhi;
    const float val = acc[v];
    support[(b * N_ + row) * F_ + col]   = val;
    supportT[(b * F_ + col) * N_ + row]  = (bf16_t)val;
  }
}

// ---------------------------------------------------------------------------
// Kernel 2: fused node pipeline for one (b, 16-row tile):
//   phase1: out1[e] = adj_t[b,e] @ support[b]  -> tanh(out1+support) -> LDS X2
//   phase2: out = tanh(X2 @ out_linear_W)      -> d_out + bf16 LDS tile
//   phase3: PQ  = out @ [W_top | W_bot]        -> workspace
// ---------------------------------------------------------------------------
__global__ __launch_bounds__(256)
void node_kernel(const float* __restrict__ lcq,
                 const float* __restrict__ support,
                 const bf16_t* __restrict__ supportT,
                 const bf16_t* __restrict__ outWt,
                 const bf16_t* __restrict__ WcatT,
                 float* __restrict__ out,          // d_out[0 .. B*N*U)
                 float* __restrict__ PQ) {
  __shared__ __attribute__((aligned(16))) bf16_t X2[16][F_ * E_];   // 32 KiB
  __shared__ __attribute__((aligned(16))) bf16_t outLds[16][U_];    //  4 KiB

  const int b    = blockIdx.x >> 3;
  const int n0   = (blockIdx.x & 7) << 4;
  const int wave = threadIdx.x >> 5;
  const int lane = threadIdx.x & 31;
  const int r    = lane & 15;
  const int kk   = (lane >> 4) << 3;
  const int mhi  = (lane >> 4) << 3;

  // ---- phase 1: wave = edge type e ---------------------------------------
  {
    const int e = wave;
    v8f acc[8];
#pragma unroll
    for (int ft = 0; ft < 8; ++ft) acc[ft] = (v8f){};

    for (int m0 = 0; m0 < N_; m0 += 32) {
      // A: adj_t[b,e][n0+r][m] = lcq[b, n0+r, m, e]  (stride E in f32)
      const float* ap = lcq + (((b * N_ + n0 + r) * N_) + m0 + kk) * E_ + e;
      v16bf a;
#pragma unroll
      for (int t = 0; t < 8; ++t) a[t]     = (bf16_t)ap[t * E_];
#pragma unroll
      for (int t = 0; t < 8; ++t) a[8 + t] = (bf16_t)ap[(16 + t) * E_];
#pragma unroll
      for (int ft = 0; ft < 8; ++ft) {
        v16bf bf = load_frag_rm(supportT + (b * F_ + (ft << 4)) * N_ + m0, N_, lane);
        acc[ft] = wmma_bf16(a, bf, acc[ft]);
      }
    }
    // self add + tanh; scatter into X2[m][f*E + e]
#pragma unroll
    for (int ft = 0; ft < 8; ++ft) {
      const int f = (ft << 4) + (lane & 15);
#pragma unroll
      for (int v = 0; v < 8; ++v) {
        const int m = v + mhi;
        const float s   = support[(b * N_ + n0 + m) * F_ + f];
        const float val = tanhf(acc[ft][v] + s);
        X2[m][f * E_ + e] = (bf16_t)val;
      }
    }
  }
  __syncthreads();

  // ---- phase 2: wave = 16-wide u tile, K = 1024 --------------------------
  {
    const int u0 = wave << 4;
    v8f c = {};
    for (int k0 = 0; k0 < F_ * E_; k0 += 32) {
      v16bf a  = load_frag_rm(&X2[0][0] + k0, F_ * E_, lane);
      v16bf bf = load_frag_rm(outWt + u0 * (F_ * E_) + k0, F_ * E_, lane);
      c = wmma_bf16(a, bf, c);
    }
    const int col = u0 + (lane & 15);
#pragma unroll
    for (int v = 0; v < 8; ++v) {
      const int m = v + mhi;
      const float val = tanhf(c[v]);
      out[(b * N_ + n0 + m) * U_ + col] = val;
      outLds[m][col] = (bf16_t)val;
    }
  }
  __syncthreads();

  // ---- phase 3: PQ = out_tile @ WcatT^T, wave = 16-wide col tile ---------
  {
    const int p0 = wave << 4;
    v8f c = {};
#pragma unroll
    for (int k0 = 0; k0 < U_; k0 += 32) {
      v16bf a  = load_frag_rm(&outLds[0][0] + k0, U_, lane);
      v16bf bf = load_frag_rm(WcatT + p0 * U_ + k0, U_, lane);
      c = wmma_bf16(a, bf, c);
    }
    const int col = p0 + (lane & 15);
#pragma unroll
    for (int v = 0; v < 8; ++v) {
      const int m = v + mhi;
      PQ[(b * N_ + n0 + m) * 128 + col] = c[v];
    }
  }
}

// ---------------------------------------------------------------------------
// Kernel 3: edge stage. Block = (b, i); wave = 16-row j tile.
//   W6t (12 KiB) is staged global->LDS with GLOBAL_LOAD_ASYNC_TO_LDS_B128
//   (ASYNCcnt), then all B-fragments come from LDS (ds_load_b128).
//   X6 = [ relu(P[b,j]+Q[b,i]) | lcq_adj[b,i,j,:] | 0pad ]  (16 x 96, bf16)
//   adj = relu(X6 @ W6)  -> d_out adj region (the 134 MB stream)
// ---------------------------------------------------------------------------
__global__ __launch_bounds__(256)
void adj_kernel(const float* __restrict__ lcq,
                const float* __restrict__ PQ,
                const bf16_t* __restrict__ W6t,
                float* __restrict__ adj_out) {
  __shared__ __attribute__((aligned(16))) bf16_t X6[8][16][K6];     // 24 KiB
  __shared__ __attribute__((aligned(16))) bf16_t W6lds[UE_ * K6];   // 12 KiB

  const int b    = blockIdx.x >> 7;
  const int i    = blockIdx.x & 127;
  const int wave = threadIdx.x >> 5;
  const int lane = threadIdx.x & 31;
  const int j0   = wave << 4;

  // --- async stage W6t -> LDS: 12 KiB = 3 rounds of 256 lanes x 16 B ------
  {
    const int tid = (int)threadIdx.x;
#pragma unroll
    for (int rnd = 0; rnd < 3; ++rnd) {
      const int byte_off = (rnd * 256 + tid) * 16;
      const unsigned lds = lds_addr_of((const char*)W6lds + byte_off);
      const unsigned long long ga =
          (unsigned long long)((const char*)W6t + byte_off);
      asm volatile("global_load_async_to_lds_b128 %0, %1, off"
                   :: "v"(lds), "v"(ga) : "memory");
    }
  }

  const float* Qrow = PQ + (b * N_ + i) * 128 + UE_;
  const float* adj_row = lcq + ((b * N_ + i) * N_ + j0) * E_;
  __builtin_prefetch(adj_row, 0, 1);             // global_prefetch_b8

  // build X6 tile (1536 elems / 32 lanes = 48 per lane)
  for (int idx = lane; idx < 16 * K6; idx += 32) {
    const int m = idx / K6, c = idx % K6;
    float v;
    if (c < UE_) {
      v = fmaxf(PQ[(b * N_ + j0 + m) * 128 + c] + Qrow[c], 0.0f);
    } else if (c < UE_ + E_) {
      v = adj_row[m * E_ + (c - UE_)];
    } else {
      v = 0.0f;
    }
    X6[wave][m][c] = (bf16_t)v;
  }

  // drain async copies, then block-wide sync so everyone sees W6lds + X6.
  asm volatile("s_wait_asynccnt 0x0" ::: "memory");
  __syncthreads();

  v8f c[4];
#pragma unroll
  for (int nt = 0; nt < 4; ++nt) c[nt] = (v8f){};
#pragma unroll
  for (int kf = 0; kf < 3; ++kf) {
    v16bf a = load_frag_rm(&X6[wave][0][0] + kf * 32, K6, lane);
#pragma unroll
    for (int nt = 0; nt < 4; ++nt) {
      v16bf bf = load_frag_rm(W6lds + (nt << 4) * K6 + kf * 32, K6, lane);
      c[nt] = wmma_bf16(a, bf, c[nt]);
    }
  }
  const int mhi = (lane >> 4) << 3;
#pragma unroll
  for (int nt = 0; nt < 4; ++nt) {
    const int col = (nt << 4) + (lane & 15);
#pragma unroll
    for (int v = 0; v < 8; ++v) {
      const int m = v + mhi;
      adj_out[((b * N_ + i) * N_ + j0 + m) * UE_ + col] = fmaxf(c[nt][v], 0.0f);
    }
  }
}

// ---------------------------------------------------------------------------
// Launch
// ---------------------------------------------------------------------------
extern "C" void kernel_launch(void* const* d_in, const int* in_sizes, int n_in,
                              void* d_out, int out_size, void* d_ws, size_t ws_size,
                              hipStream_t stream) {
  const float* nf      = (const float*)d_in[0];   // [B,N,F]
  const float* lcq     = (const float*)d_in[1];   // [B,N,N,E]
  const float* testW   = (const float*)d_in[2];   // [F,F]
  const float* outW    = (const float*)d_in[3];   // [F*E,U]
  const float* adjW    = (const float*)d_in[4];   // [2U,UE]
  const float* adjOutW = (const float*)d_in[5];   // [UE+E,UE]

  char* ws = (char*)d_ws;
  float*  support  = (float*) (ws + WS_SUPPORT);
  bf16_t* supportT = (bf16_t*)(ws + WS_SUPPORT_T);
  float*  PQ       = (float*) (ws + WS_PQ);
  bf16_t* testWt   = (bf16_t*)(ws + WS_TESTWT);
  bf16_t* outWt    = (bf16_t*)(ws + WS_OUTWT);
  bf16_t* WcatT    = (bf16_t*)(ws + WS_WCATT);
  bf16_t* W6t      = (bf16_t*)(ws + WS_W6T);

  float* out_node = (float*)d_out;                       // [B,N,U]
  float* out_adj  = out_node + B_ * N_ * U_;             // [B,N,N,UE]

  const int prep_elems = F_ * F_ + U_ * F_ * E_ + 128 * 128 + UE_ * K6;
  prep_weights<<<(prep_elems + 255) / 256, 256, 0, stream>>>(
      testW, outW, adjW, adjOutW, testWt, outWt, WcatT, W6t);

  support_kernel<<<B_ * (N_ / 16), 256, 0, stream>>>(nf, testWt, support, supportT);

  node_kernel<<<B_ * (N_ / 16), 256, 0, stream>>>(lcq, support, supportT, outWt,
                                                  WcatT, out_node, PQ);

  adj_kernel<<<B_ * N_, 256, 0, stream>>>(lcq, PQ, W6t, out_adj);
}
